// my_multihead_self_attention_66408784331368
// MI455X (gfx1250) — matrix operand order repaired and checked by reading, not compiled, not run on hardware
//
#include <hip/hip_runtime.h>
#include <hip/hip_bf16.h>
#include <stdint.h>

// MI455X / gfx1250: wave32, WMMA 16x16x32 bf16 (f32 accumulate),
// async global->LDS staging for GEMM weight tiles.

typedef __attribute__((ext_vector_type(16))) __bf16 v16bf;
typedef __attribute__((ext_vector_type(8)))  __bf16 v8bf;
typedef __attribute__((ext_vector_type(8)))  float  v8f;

#define D_MODEL 1024
#define N_HEADS 16
#define DH      64
#define SEQ     2048
#define BATCH   4
#define KSLICE  64

__device__ __forceinline__ __bf16 f2bf(float f) {
  unsigned u = __builtin_bit_cast(unsigned, f);
  u += 0x7fffu + ((u >> 16) & 1u);             // round-to-nearest-even
  unsigned short s = (unsigned short)(u >> 16);
  return __builtin_bit_cast(__bf16, s);
}

// load 8 contiguous fp32 -> 8 bf16
__device__ __forceinline__ v8bf ldcvt8(const float* p) {
  float4 a = *(const float4*)p;
  float4 b = *(const float4*)(p + 4);
  v8bf r;
  r[0]=f2bf(a.x); r[1]=f2bf(a.y); r[2]=f2bf(a.z); r[3]=f2bf(a.w);
  r[4]=f2bf(b.x); r[5]=f2bf(b.y); r[6]=f2bf(b.z); r[7]=f2bf(b.w);
  return r;
}

union BF16x16 { v16bf v; v8bf h[2]; };

#define WMMA_BF16(A, B, C) __builtin_amdgcn_wmma_f32_16x16x32_bf16( \
    false, (A), false, (B), (short)0, (C), false, false)

// async copy of one 16B chunk (per lane) from global to LDS; ASYNCcnt-tracked
__device__ __forceinline__ void async_b128(uint32_t lds_off, const __bf16* g) {
  asm volatile("global_load_async_to_lds_b128 %0, %1, off"
               :: "v"(lds_off), "v"(g) : "memory");
}
__device__ __forceinline__ void wait_async0() {
  asm volatile("s_wait_asynccnt 0x0" ::: "memory");
}

// ---------------------------------------------------------------------------
// Kernel 0: one-shot fp32 -> bf16 conversion (8 elements / thread)
// ---------------------------------------------------------------------------
__global__ __launch_bounds__(256)
void cvt_bf16_k(const float* __restrict__ src, __bf16* __restrict__ dst, int n8) {
  const int i = blockIdx.x * 256 + threadIdx.x;
  if (i < n8) {
    *(v8bf*)(dst + (size_t)i * 8) = ldcvt8(src + (size_t)i * 8);
  }
}

// ---------------------------------------------------------------------------
// Kernel 1: xqkv = x @ w_qkv^T (bf16), fused split into Q(scaled), K
//           ([B,H,S,64]) and V^T ([B,H,64,S]).
// Weight tile (64 rows x 64 k) staged in LDS via async copy, double-buffered.
// A-frag (x row):  lane M = lane&15; elems K = 8*hh+i (i<8), 16+8*hh+(i-8)
// B-frag (w row):  lane N = lane&15; elems K = 8*hh+i (i<8), 16+8*hh+(i-8)
// ---------------------------------------------------------------------------
__global__ __launch_bounds__(256)
void qkv_gemm_k(const __bf16* __restrict__ Xb, const __bf16* __restrict__ Wb,
                __bf16* __restrict__ Qb, __bf16* __restrict__ Kb,
                __bf16* __restrict__ Vt) {
  __shared__ __bf16 wtile[2][64][KSLICE];          // 2 x 8KB

  const int lane = threadIdx.x & 31;
  const int wave = threadIdx.x >> 5;
  const int hh = lane >> 4;
  const int ln = lane & 15;
  const int nbase = blockIdx.x * 64;                 // over 3*D_MODEL
  const int mbase = blockIdx.y * 256 + wave * 32;    // over B*S

  const __bf16* arow0 = Xb + (size_t)(mbase + ln)      * D_MODEL;
  const __bf16* arow1 = Xb + (size_t)(mbase + 16 + ln) * D_MODEL;

  // stage weight slice ks into buffer buf (whole workgroup, 2 chunks/thread)
  auto stage = [&](int ks, int buf) {
    #pragma unroll
    for (int c = 0; c < 2; ++c) {
      const int chunk = threadIdx.x + c * 256;       // 0..511
      const int row = chunk >> 3;                    // 0..63
      const int col = (chunk & 7) * 8;               // 0..56
      const __bf16* g = Wb + (size_t)(nbase + row) * D_MODEL + ks * KSLICE + col;
      async_b128((uint32_t)(uintptr_t)&wtile[buf][row][col], g);
    }
  };

  v8f acc[2][4];
  #pragma unroll
  for (int u = 0; u < 2; ++u)
    #pragma unroll
    for (int t = 0; t < 4; ++t)
      #pragma unroll
      for (int r = 0; r < 8; ++r) acc[u][t][r] = 0.0f;

  stage(0, 0);
  wait_async0();
  __syncthreads();

  for (int ks = 0; ks < D_MODEL / KSLICE; ++ks) {
    const int buf = ks & 1;
    if (ks + 1 < D_MODEL / KSLICE) stage(ks + 1, buf ^ 1);

    #pragma unroll
    for (int kh = 0; kh < 2; ++kh) {
      const int kk = ks * KSLICE + kh * 32;
      BF16x16 a0, a1, bm[4];
      a0.h[0] = *(const v8bf*)(arow0 + kk + 8*hh);
      a0.h[1] = *(const v8bf*)(arow0 + kk + 16 + 8*hh);
      a1.h[0] = *(const v8bf*)(arow1 + kk + 8*hh);
      a1.h[1] = *(const v8bf*)(arow1 + kk + 16 + 8*hh);
      #pragma unroll
      for (int t = 0; t < 4; ++t) {
        bm[t].h[0] = *(const v8bf*)&wtile[buf][16*t + ln][kh*32 + 8*hh];
        bm[t].h[1] = *(const v8bf*)&wtile[buf][16*t + ln][kh*32 + 16 + 8*hh];
      }
      #pragma unroll
      for (int t = 0; t < 4; ++t) {
        acc[0][t] = WMMA_BF16(a0.v, bm[t].v, acc[0][t]);
        acc[1][t] = WMMA_BF16(a1.v, bm[t].v, acc[1][t]);
      }
    }
    wait_async0();        // next buffer filled
    __syncthreads();      // all waves done reading current buffer
  }

  // epilogue: whole 64-wide N block lies in one region & one head
  const int region = nbase / D_MODEL;              // 0=Q 1=K 2=V
  const int head   = (nbase % D_MODEL) / DH;
  #pragma unroll
  for (int u = 0; u < 2; ++u) {
    #pragma unroll
    for (int t = 0; t < 4; ++t) {
      const int d = 16*t + ln;                     // col within head
      #pragma unroll
      for (int r = 0; r < 8; ++r) {
        const int m = mbase + 16*u + 8*hh + r;
        const int b = m >> 11, s = m & (SEQ - 1);
        const float val = acc[u][t][r];
        if (region == 0)
          Qb[((size_t)(b*N_HEADS + head)*SEQ + s)*DH + d] = f2bf(val * 0.125f);
        else if (region == 1)
          Kb[((size_t)(b*N_HEADS + head)*SEQ + s)*DH + d] = f2bf(val);
        else
          Vt[((size_t)(b*N_HEADS + head)*DH + d)*SEQ + s] = f2bf(val);
      }
    }
  }
}

// ---------------------------------------------------------------------------
// Kernel 2: causal flash attention. One wave owns 32 q rows (2 q tiles) so
// each K/V fragment feeds two WMMAs.
// S^T = K (A) x Q^T (B): C-layout lane holds qi=lane&15, elems kj=8*hh+r
// O^T = V^T (A) x P^T (B): P^T assembled from S^T via shfl_xor(16)
// ---------------------------------------------------------------------------
__global__ __launch_bounds__(256)
void attn_k(const __bf16* __restrict__ Qb, const __bf16* __restrict__ Kb,
            const __bf16* __restrict__ Vt, __bf16* __restrict__ Ob) {
  const int lane = threadIdx.x & 31;
  const int wave = threadIdx.x >> 5;
  const int hh = lane >> 4;
  const int ln = lane & 15;
  const int head = blockIdx.y;
  const int b    = blockIdx.z;
  const int qbase = blockIdx.x * 256 + wave * 32;    // multiple of 32

  const __bf16* Qh = Qb + (size_t)(b*N_HEADS + head) * SEQ * DH;
  const __bf16* Kh = Kb + (size_t)(b*N_HEADS + head) * SEQ * DH;
  const __bf16* Vh = Vt + (size_t)(b*N_HEADS + head) * DH * SEQ;

  // Q as B operand: lane holds column qi, d = 32*dt + 8*hh + i (i<8), 32*dt + 16 + 8*hh + (i-8)
  BF16x16 qf[2][2];
  int qglob[2];
  #pragma unroll
  for (int qt = 0; qt < 2; ++qt) {
    qglob[qt] = qbase + 16*qt + ln;
    const __bf16* qrow = Qh + (size_t)qglob[qt] * DH;
    #pragma unroll
    for (int dt = 0; dt < 2; ++dt) {
      qf[qt][dt].h[0] = *(const v8bf*)(qrow + 32*dt + 8*hh);
      qf[qt][dt].h[1] = *(const v8bf*)(qrow + 32*dt + 16 + 8*hh);
    }
  }

  v8f acc[2][4];
  #pragma unroll
  for (int qt = 0; qt < 2; ++qt)
    #pragma unroll
    for (int t = 0; t < 4; ++t)
      #pragma unroll
      for (int r = 0; r < 8; ++r) acc[qt][t][r] = 0.0f;
  float m_run[2] = {-3.0e38f, -3.0e38f};
  float l_run[2] = {0.0f, 0.0f};

  const int kend = qbase + 32;                 // keys kj <= qbase+31
  for (int k0 = 0; k0 < kend; k0 += 32) {
    // ---- scores^T: tiles [k0,k0+16) and [k0+16,k0+32) for both q tiles
    v8f s[2][2];
    #pragma unroll
    for (int qt = 0; qt < 2; ++qt)
      #pragma unroll
      for (int r = 0; r < 8; ++r) { s[qt][0][r] = 0.0f; s[qt][1][r] = 0.0f; }
    #pragma unroll
    for (int dt = 0; dt < 2; ++dt) {
      const __bf16* kr0 = Kh + (size_t)(k0 + ln)      * DH + 32*dt;
      const __bf16* kr1 = Kh + (size_t)(k0 + 16 + ln) * DH + 32*dt;
      BF16x16 kf0, kf1;
      kf0.h[0] = *(const v8bf*)(kr0 + 8*hh);
      kf0.h[1] = *(const v8bf*)(kr0 + 16 + 8*hh);
      kf1.h[0] = *(const v8bf*)(kr1 + 8*hh);
      kf1.h[1] = *(const v8bf*)(kr1 + 16 + 8*hh);
      #pragma unroll
      for (int qt = 0; qt < 2; ++qt) {
        s[qt][0] = WMMA_BF16(kf0.v, qf[qt][dt].v, s[qt][0]);
        s[qt][1] = WMMA_BF16(kf1.v, qf[qt][dt].v, s[qt][1]);
      }
    }

    // ---- per-q-tile online softmax; only last chunk crosses the diagonal
    const bool need_mask = (k0 + 32 > qbase);
    BF16x16 pf[2];
    #pragma unroll
    for (int qt = 0; qt < 2; ++qt) {
      float v0[8], v1[8], mc = -3.0e38f;
      #pragma unroll
      for (int r = 0; r < 8; ++r) {
        const int kj0 = k0 + 8*hh + r;
        const int kj1 = k0 + 16 + 8*hh + r;
        v0[r] = (need_mask && kj0 > qglob[qt]) ? -3.0e38f : s[qt][0][r];
        v1[r] = (need_mask && kj1 > qglob[qt]) ? -3.0e38f : s[qt][1][r];
        mc = fmaxf(mc, fmaxf(v0[r], v1[r]));
      }
      mc = fmaxf(mc, __shfl_xor(mc, 16, 32));
      const float m_new = fmaxf(m_run[qt], mc);
      const float scale = __expf(m_run[qt] - m_new);
      float p0[8], p1[8], lsum = 0.0f;
      #pragma unroll
      for (int r = 0; r < 8; ++r) {
        p0[r] = __expf(v0[r] - m_new);
        p1[r] = __expf(v1[r] - m_new);
        lsum += p0[r] + p1[r];
      }
      lsum += __shfl_xor(lsum, 16, 32);
      l_run[qt] = l_run[qt] * scale + lsum;
      m_run[qt] = m_new;
      #pragma unroll
      for (int t = 0; t < 4; ++t)
        #pragma unroll
        for (int r = 0; r < 8; ++r) acc[qt][t][r] *= scale;

      // P^T as B operand: slot i<8 -> kj=16*hh+i, slot i>=8 -> 16*hh+8+(i-8)
      #pragma unroll
      for (int r = 0; r < 8; ++r) {
        const float o0 = __shfl_xor(p0[r], 16, 32);   // partner's tile0
        const float o1 = __shfl_xor(p1[r], 16, 32);   // partner's tile1
        const float lo = hh ? o1 : p0[r];
        const float hi = hh ? p1[r] : o0;
        pf[qt].h[0][r] = f2bf(lo);
        pf[qt].h[1][r] = f2bf(hi);
      }
    }

    // ---- O^T += V^T(tile t) x P^T : V frag shared by both q tiles
    #pragma unroll
    for (int t = 0; t < 4; ++t) {
      const __bf16* vrow = Vh + (size_t)(16*t + ln) * SEQ + k0;
      BF16x16 vf;
      vf.h[0] = *(const v8bf*)(vrow + 16*hh);
      vf.h[1] = *(const v8bf*)(vrow + 16*hh + 8);
      acc[0][t] = WMMA_BF16(vf.v, pf[0].v, acc[0][t]);
      acc[1][t] = WMMA_BF16(vf.v, pf[1].v, acc[1][t]);
    }
  }

  // ---- write O rows (bf16), cols head*64 + 16*t + 8*hh + r
  #pragma unroll
  for (int qt = 0; qt < 2; ++qt) {
    const float inv_l = 1.0f / l_run[qt];
    __bf16* orow = Ob + ((size_t)b * SEQ + qglob[qt]) * D_MODEL + head * DH;
    #pragma unroll
    for (int t = 0; t < 4; ++t) {
      v8bf ov;
      #pragma unroll
      for (int r = 0; r < 8; ++r) ov[r] = f2bf(acc[qt][t][r] * inv_l);
      *(v8bf*)(orow + 16*t + 8*hh) = ov;
    }
  }
}

// ---------------------------------------------------------------------------
// Kernel 3: out = O @ w_out^T (bf16 operands, fp32 result), LDS-staged B
// ---------------------------------------------------------------------------
__global__ __launch_bounds__(256)
void out_gemm_k(const __bf16* __restrict__ Ob, const __bf16* __restrict__ Wb,
                float* __restrict__ out) {
  __shared__ __bf16 wtile[2][64][KSLICE];

  const int lane = threadIdx.x & 31;
  const int wave = threadIdx.x >> 5;
  const int hh = lane >> 4;
  const int ln = lane & 15;
  const int nbase = blockIdx.x * 64;
  const int mbase = blockIdx.y * 256 + wave * 32;

  const __bf16* arow0 = Ob + (size_t)(mbase + ln)      * D_MODEL;
  const __bf16* arow1 = Ob + (size_t)(mbase + 16 + ln) * D_MODEL;

  auto stage = [&](int ks, int buf) {
    #pragma unroll
    for (int c = 0; c < 2; ++c) {
      const int chunk = threadIdx.x + c * 256;
      const int row = chunk >> 3;
      const int col = (chunk & 7) * 8;
      const __bf16* g = Wb + (size_t)(nbase + row) * D_MODEL + ks * KSLICE + col;
      async_b128((uint32_t)(uintptr_t)&wtile[buf][row][col], g);
    }
  };

  v8f acc[2][4];
  #pragma unroll
  for (int u = 0; u < 2; ++u)
    #pragma unroll
    for (int t = 0; t < 4; ++t)
      #pragma unroll
      for (int r = 0; r < 8; ++r) acc[u][t][r] = 0.0f;

  stage(0, 0);
  wait_async0();
  __syncthreads();

  for (int ks = 0; ks < D_MODEL / KSLICE; ++ks) {
    const int buf = ks & 1;
    if (ks + 1 < D_MODEL / KSLICE) stage(ks + 1, buf ^ 1);

    #pragma unroll
    for (int kh = 0; kh < 2; ++kh) {
      const int kk = ks * KSLICE + kh * 32;
      BF16x16 a0, a1, bm[4];
      a0.h[0] = *(const v8bf*)(arow0 + kk + 8*hh);
      a0.h[1] = *(const v8bf*)(arow0 + kk + 16 + 8*hh);
      a1.h[0] = *(const v8bf*)(arow1 + kk + 8*hh);
      a1.h[1] = *(const v8bf*)(arow1 + kk + 16 + 8*hh);
      #pragma unroll
      for (int t = 0; t < 4; ++t) {
        bm[t].h[0] = *(const v8bf*)&wtile[buf][16*t + ln][kh*32 + 8*hh];
        bm[t].h[1] = *(const v8bf*)&wtile[buf][16*t + ln][kh*32 + 16 + 8*hh];
      }
      #pragma unroll
      for (int t = 0; t < 4; ++t) {
        acc[0][t] = WMMA_BF16(a0.v, bm[t].v, acc[0][t]);
        acc[1][t] = WMMA_BF16(a1.v, bm[t].v, acc[1][t]);
      }
    }
    wait_async0();
    __syncthreads();
  }

  #pragma unroll
  for (int u = 0; u < 2; ++u) {
    #pragma unroll
    for (int t = 0; t < 4; ++t) {
      const int n = nbase + 16*t + ln;
      #pragma unroll
      for (int r = 0; r < 8; ++r) {
        const int m = mbase + 16*u + 8*hh + r;
        out[(size_t)m * D_MODEL + n] = acc[u][t][r];
      }
    }
  }
}

// ---------------------------------------------------------------------------
extern "C" void kernel_launch(void* const* d_in, const int* in_sizes, int n_in,
                              void* d_out, int out_size, void* d_ws, size_t ws_size,
                              hipStream_t stream) {
  const float* x    = (const float*)d_in[0];
  const float* wqkv = (const float*)d_in[1];
  const float* wout = (const float*)d_in[2];
  float* out = (float*)d_out;

  // workspace layout (bf16 elements):
  const size_t XN   = (size_t)BATCH * SEQ * D_MODEL;        // 8Mi
  const size_t WQN  = (size_t)3 * D_MODEL * D_MODEL;        // 3Mi
  const size_t WON  = (size_t)D_MODEL * D_MODEL;            // 1Mi
  const size_t QKVN = (size_t)BATCH * N_HEADS * SEQ * DH;   // 8Mi

  __bf16* Xbuf = (__bf16*)d_ws;
  __bf16* Wqkb = Xbuf + XN;
  __bf16* Wotb = Wqkb + WQN;
  __bf16* Qb   = Wotb + WON;
  __bf16* Kb   = Qb + QKVN;
  __bf16* Vt   = Kb + QKVN;
  __bf16* Ob   = Vt + QKVN;

  dim3 blk(256);

  // one-shot bf16 conversions
  cvt_bf16_k<<<dim3((unsigned)(XN / 8 / 256)),  blk, 0, stream>>>(x,    Xbuf, (int)(XN / 8));
  cvt_bf16_k<<<dim3((unsigned)(WQN / 8 / 256)), blk, 0, stream>>>(wqkv, Wqkb, (int)(WQN / 8));
  cvt_bf16_k<<<dim3((unsigned)(WON / 8 / 256)), blk, 0, stream>>>(wout, Wotb, (int)(WON / 8));

  dim3 g1(3 * D_MODEL / 64, BATCH * SEQ / 256);   // (48, 32)
  qkv_gemm_k<<<g1, blk, 0, stream>>>(Xbuf, Wqkb, Qb, Kb, Vt);

  dim3 g2(SEQ / 256, N_HEADS, BATCH);             // (8, 16, 4)
  attn_k<<<g2, blk, 0, stream>>>(Qb, Kb, Vt, Ob);

  dim3 g3(D_MODEL / 64, BATCH * SEQ / 256);       // (16, 32)
  out_gemm_k<<<g3, blk, 0, stream>>>(Ob, Wotb, out);
}
